// MultiHeadAttentionWithRoPE_26199300505997
// MI455X (gfx1250) — compile-verified, matrix operand order checked by reading, not run on hardware
//
#include <hip/hip_runtime.h>
#include <hip/hip_bf16.h>

// ---------------------------------------------------------------------------
// MultiHeadAttention + RoPE for MI455X (gfx1250), wave32, WMMA bf16 path.
// B=2, S=2048, D=1024, H=16, dk=64. fp32 in/out; bf16 WMMA operands with
// fp32 accumulation (v_wmma_f32_16x16x32_bf16).
// Attention: 64-key blocks, block-cooperative double-buffered async LDS
// staging (GLOBAL_LOAD_ASYNC_TO_LDS_B128), DPP16 row_xmask softmax
// reductions (no DS-pipe shuffles), diagonal-only causal masking.
// ---------------------------------------------------------------------------

typedef __attribute__((ext_vector_type(16))) __bf16 v16bf;
typedef __attribute__((ext_vector_type(8)))  __bf16 v8bf;
typedef __attribute__((ext_vector_type(4)))  __bf16 v4bf;
typedef __attribute__((ext_vector_type(8)))  float  v8f;
typedef __attribute__((ext_vector_type(4)))  int    v4i;

#define B_  2
#define S_  2048
#define D_  1024
#define H_  16
#define DK_ 64

#define KB_   64   // keys per attention block-iteration
#define TROW_ 72   // padded LDS row length (64 + 8) -> 144B stride, b128-clean

#if defined(__has_builtin)
#if __has_builtin(__builtin_amdgcn_global_load_async_to_lds_b128)
#define ASYNC_LDS 1
#endif
#if __has_builtin(__builtin_amdgcn_s_wait_asynccnt)
#define HAVE_WAIT_ASYNC 1
#endif
#endif

#ifdef ASYNC_LDS
typedef v4i __attribute__((address_space(1)))* g4p;
typedef v4i __attribute__((address_space(3)))* l4p;
#endif

// Stage 16 bytes global -> LDS (async DMA when available).
__device__ inline void stage16(__bf16* l, const __bf16* g) {
#ifdef ASYNC_LDS
  __builtin_amdgcn_global_load_async_to_lds_b128((g4p)g, (l4p)l, 0, 0);
#else
  *(v8bf*)l = *(const v8bf*)g;
#endif
}

__device__ inline void stage_fence() {
#if defined(ASYNC_LDS) && defined(HAVE_WAIT_ASYNC)
  __builtin_amdgcn_s_wait_asynccnt(0);
#endif
  __syncthreads();
}

__device__ inline v8f wmma_bf16(v16bf a, v16bf b, v8f c) {
  return __builtin_amdgcn_wmma_f32_16x16x32_bf16(false, a, false, b, (short)0, c,
                                                 false, false);
}

__device__ inline v8f vzero8() {
  v8f z;
#pragma unroll
  for (int j = 0; j < 8; ++j) z[j] = 0.0f;
  return z;
}

__device__ inline v8f vsplat8(float x) {
  v8f z;
#pragma unroll
  for (int j = 0; j < 8; ++j) z[j] = x;
  return z;
}

__device__ inline v16bf cat8(v8bf lo, v8bf hi) {
  return __builtin_shufflevector(lo, hi, 0, 1, 2, 3, 4, 5, 6, 7, 8, 9, 10, 11,
                                 12, 13, 14, 15);
}

// WMMA 16x16x32 bf16 fragment load, per ISA layout: lanes 0-15 hold K in
// [0,8)+[16,24), lanes 16-31 hold K in [8,16)+[24,32). Caller passes
// ptr = row_base + khalf (khalf = (lane>>4)*8). Two 16B vector loads.
__device__ inline v16bf load_frag(const __bf16* p) {
  const v8bf lo = *(const v8bf*)p;
  const v8bf hi = *(const v8bf*)(p + 16);
  return cat8(lo, hi);
}

// DPP16 ROW_XMASK lane-xor within each 16-lane row: ctrl = 0x160 | mask.
template <int MASK>
__device__ inline float dpp_xor16(float x) {
  return __builtin_bit_cast(
      float, __builtin_amdgcn_update_dpp(0, __builtin_bit_cast(int, x),
                                         0x160 | MASK, 0xf, 0xf, true));
}

// Row reductions inside a 16-lane half-wave, VALU-only (no DS pipe).
__device__ inline float hg_max(float v) {
  v = fmaxf(v, dpp_xor16<1>(v));
  v = fmaxf(v, dpp_xor16<2>(v));
  v = fmaxf(v, dpp_xor16<4>(v));
  v = fmaxf(v, dpp_xor16<8>(v));
  return v;
}
__device__ inline float hg_sum(float v) {
  v += dpp_xor16<1>(v);
  v += dpp_xor16<2>(v);
  v += dpp_xor16<4>(v);
  v += dpp_xor16<8>(v);
  return v;
}

// ---------------------------------------------------------------------------
// fp32 -> bf16 bulk convert (4 elements / thread, vectorized)
// ---------------------------------------------------------------------------
__global__ __launch_bounds__(256) void f32_to_bf16_kernel(
    const float* __restrict__ in, __bf16* __restrict__ out, int n4) {
  const int i = (int)(blockIdx.x * blockDim.x + threadIdx.x);
  if (i >= n4) return;
  const float4 f = ((const float4*)in)[i];
  v4bf o;
  o[0] = (__bf16)f.x;
  o[1] = (__bf16)f.y;
  o[2] = (__bf16)f.z;
  o[3] = (__bf16)f.w;
  ((v4bf*)out)[i] = o;
}

// ---------------------------------------------------------------------------
// GEMM: C[M,N] = A[M,K] * W[N,K]^T + bias[N], bf16 operands.
// One wave -> 32x64 strip (8 WMMAs per 32-deep k step).
// VTSTORE: write result as bf16 to V^T layout [B*H][dk][S] (for attention).
// ---------------------------------------------------------------------------
template <typename TOut, bool VTSTORE>
__global__ __launch_bounds__(256) void gemm_bias_kernel(
    const __bf16* __restrict__ A, const __bf16* __restrict__ W,
    const float* __restrict__ bias, TOut* __restrict__ C, int M, int N, int K) {
  const int wave    = (int)((blockIdx.x * blockDim.x + threadIdx.x) >> 5);
  const int lane    = (int)(threadIdx.x & 31);
  const int nGroups = N >> 6;  // 64 cols per wave
  const int mTile   = wave / nGroups;
  const int nGroup  = wave - mTile * nGroups;
  if (mTile * 32 >= M) return;  // wave-uniform

  const int m16   = lane & 15;
  const int khalf = (lane >> 4) << 3;  // 0 or 8
  const int nBase = nGroup * 64;
  const __bf16* arow0 = A + (size_t)(mTile * 32 + m16) * (size_t)K + khalf;
  const __bf16* arow1 = arow0 + (size_t)16 * K;

  v8f acc[2][4];
#pragma unroll
  for (int r = 0; r < 2; ++r)
#pragma unroll
    for (int c = 0; c < 4; ++c) acc[r][c] = vzero8();

  for (int k = 0; k < K; k += 32) {
    if (k + 32 < K) {
      __builtin_prefetch(arow0 + k + 32, 0, 1);
      __builtin_prefetch(arow1 + k + 32, 0, 1);
    }
    const v16bf a0 = load_frag(arow0 + k);
    const v16bf a1 = load_frag(arow1 + k);
#pragma unroll
    for (int c = 0; c < 4; ++c) {
      const __bf16* wrow =
          W + (size_t)(nBase + c * 16 + m16) * (size_t)K + khalf;
      const v16bf b = load_frag(wrow + k);
      acc[0][c] = wmma_bf16(a0, b, acc[0][c]);
      acc[1][c] = wmma_bf16(a1, b, acc[1][c]);
    }
  }

  // C layout: vgpr j, lane L -> row = j + 8*(L>=16), col = L&15.
#pragma unroll
  for (int r = 0; r < 2; ++r) {
#pragma unroll
    for (int c = 0; c < 4; ++c) {
      const int col = nBase + c * 16 + m16;
      const float bv = bias[col];
#pragma unroll
      for (int j = 0; j < 8; ++j) {
        const int row = mTile * 32 + r * 16 + j + khalf;
        const float val = acc[r][c][j] + bv;
        if constexpr (VTSTORE) {
          // V^T: [b*H + h][cc][s]; row = b*S + s, col = h*64 + cc
          const int bb = row >> 11;  // / S_
          const int ss = row & (S_ - 1);
          const int hh = col >> 6;
          const int cc = col & (DK_ - 1);
          ((__bf16*)C)[(((size_t)bb * H_ + hh) * DK_ + cc) * S_ + ss] =
              (__bf16)val;
        } else {
          C[(size_t)row * (size_t)N + col] = (TOut)val;
        }
      }
    }
  }
}

// ---------------------------------------------------------------------------
// RoPE (exact reference semantics: repeat_interleave freqs + rotate_half),
// fold 1/sqrt(dk) into Q, convert Q/K to bf16.
// ---------------------------------------------------------------------------
__global__ __launch_bounds__(256) void rope_convert_kernel(
    const float* __restrict__ Qf, const float* __restrict__ Kf,
    __bf16* __restrict__ Qb, __bf16* __restrict__ Kb) {
  const size_t idx = (size_t)blockIdx.x * blockDim.x + threadIdx.x;
  const int c  = (int)(idx & (D_ - 1));
  const int s  = (int)((idx >> 10) & (S_ - 1));  // D_ == 1024 == 1<<10
  const int cc = c & (DK_ - 1);

  const float ex  = (float)(cc & ~1) / (float)DK_;
  const float inv = __expf(-ex * 9.210340371976184f);  // ln(10000)
  float sn, cs;
  sincosf((float)s * inv, &sn, &cs);

  const size_t partner = (cc < 32) ? idx + 32 : idx - 32;
  const float  sign    = (cc < 32) ? -1.0f : 1.0f;

  const float q = Qf[idx] * cs + sign * Qf[partner] * sn;
  const float k = Kf[idx] * cs + sign * Kf[partner] * sn;
  Qb[idx] = (__bf16)(q * 0.125f);  // fold 1/sqrt(64)
  Kb[idx] = (__bf16)k;
}

// ---------------------------------------------------------------------------
// Flash attention. One block = 8 waves = 8 consecutive 16-row q-tiles of one
// (b,h). The block streams causal key blocks of 64, cooperatively staging
// K (64x64) and V^T (64x64) tiles into double-buffered LDS (async DMA).
// Per wave per block: up to 8 WMMAs for scores, online softmax with DPP16
// reductions, P relayout through bf16 LDS, 8 WMMAs for P*V.
// Only the diagonal key block is masked; fully-masked 16-key subtiles of the
// diagonal block skip their WMMAs (wave-uniform).
// ---------------------------------------------------------------------------
__global__ __launch_bounds__(256) void flash_attn_kernel(
    const __bf16* __restrict__ Q, const __bf16* __restrict__ K,
    const __bf16* __restrict__ Vt, __bf16* __restrict__ O) {
  __shared__ __align__(16) __bf16 kbuf[2][64 * TROW_];  // keys x dk
  __shared__ __align__(16) __bf16 vbuf[2][64 * TROW_];  // dk x keys
  __shared__ __align__(16) __bf16 plds[8][16 * TROW_];  // q x keys (per wave)

  const int tid    = (int)threadIdx.x;
  const int waveId = tid >> 5;
  const int lane   = tid & 31;
  const int bh = (int)blockIdx.y;  // b*H + h
  const int b = bh >> 4, h = bh & 15;
  const int qt    = (int)blockIdx.x * 8 + waveId;
  const int qBase = qt * 16;
  const int m16   = lane & 15;
  const int khalf = (lane >> 4) << 3;  // doubles as row-half offset

  const __bf16* Qp  = Q + (size_t)b * S_ * D_ + (size_t)h * DK_;
  const __bf16* Kp  = K + (size_t)b * S_ * D_ + (size_t)h * DK_;
  const __bf16* Vtp = Vt + (size_t)bh * DK_ * S_;  // [dk][S]

  // Q A-fragments for dk slices [0,32) and [32,64); Q pre-scaled by 1/8.
  const __bf16* qrow = Qp + (size_t)(qBase + m16) * D_ + khalf;
  v16bf qa[2];
  qa[0] = load_frag(qrow);
  qa[1] = load_frag(qrow + 32);

  float mstat[8], lstat[8];
#pragma unroll
  for (int j = 0; j < 8; ++j) {
    mstat[j] = -__builtin_inff();
    lstat[j] = 0.0f;
  }
  v8f oacc[4];
#pragma unroll
  for (int c = 0; c < 4; ++c) oacc[c] = vzero8();

  __bf16* my = &plds[waveId][0];

  const int myFull = qBase >> 6;  // fully-unmasked 64-key blocks
  // Block-uniform trip count (last wave's diagonal block):
  const int nBlocksBlk = ((((int)blockIdx.x * 8 + 7) * 16) >> 6) + 1;

  // Cooperative staging of one 64-key K/V block: 2 passes x 256 thr x 16B.
  auto stage_tiles = [&](int bi2, int keyBase2) {
#pragma unroll
    for (int p = 0; p < 2; ++p) {
      const int r  = (tid >> 3) + p * 32;  // 0..63 (key row / channel row)
      const int ck = (tid & 7) << 3;       // 0..56
      stage16(&kbuf[bi2][r * TROW_ + ck],
              Kp + (size_t)(keyBase2 + r) * D_ + ck);
      stage16(&vbuf[bi2][r * TROW_ + ck],
              Vtp + (size_t)r * S_ + keyBase2 + ck);
    }
  };

  stage_tiles(0, 0);
  stage_fence();

  for (int jb = 0; jb < nBlocksBlk; ++jb) {
    const int bi = jb & 1;
    if (jb + 1 < nBlocksBlk) stage_tiles(bi ^ 1, (jb + 1) * KB_);

    if (jb <= myFull) {  // wave-uniform
      const int keyBase = jb * KB_;
      // Last 16-key subtile with any unmasked element (wave-uniform).
      const int tMax = (jb == myFull) ? ((qBase + 15 - keyBase) >> 4) : 3;

      // ---- scores S[16q x 64k] as four 16x16 C tiles (from LDS K tile) ---
      v8f sc[4];
#pragma unroll
      for (int t = 0; t < 4; ++t) {
        if (t <= tMax) {
          const __bf16* krow = &kbuf[bi][(t * 16 + m16) * TROW_ + khalf];
          sc[t] = wmma_bf16(qa[0], load_frag(krow), vzero8());
          sc[t] = wmma_bf16(qa[1], load_frag(krow + 32), sc[t]);
        } else {
          sc[t] = vsplat8(-__builtin_inff());
        }
      }

      // ---- causal mask: only the diagonal block --------------------------
      if (jb == myFull) {
#pragma unroll
        for (int t = 0; t < 4; ++t) {
          if (t <= tMax) {
#pragma unroll
            for (int j = 0; j < 8; ++j) {
              const int qg = qBase + j + khalf;
              const int kg = keyBase + t * 16 + m16;
              if (kg > qg) sc[t][j] = -__builtin_inff();
            }
          }
        }
      }

      // ---- online softmax (DPP16 reductions, VALU-only) ------------------
      float alpha[8];
#pragma unroll
      for (int j = 0; j < 8; ++j) {
        const float rm = hg_max(fmaxf(fmaxf(sc[0][j], sc[1][j]),
                                      fmaxf(sc[2][j], sc[3][j])));
        const float mnew = fmaxf(mstat[j], rm);
        alpha[j] = __expf(mstat[j] - mnew);
        mstat[j] = mnew;
        sc[0][j] = __expf(sc[0][j] - mnew);
        sc[1][j] = __expf(sc[1][j] - mnew);
        sc[2][j] = __expf(sc[2][j] - mnew);
        sc[3][j] = __expf(sc[3][j] - mnew);
        const float rs =
            hg_sum((sc[0][j] + sc[1][j]) + (sc[2][j] + sc[3][j]));
        lstat[j] = lstat[j] * alpha[j] + rs;
      }
#pragma unroll
      for (int c = 0; c < 4; ++c)
#pragma unroll
        for (int j = 0; j < 8; ++j) oacc[c][j] *= alpha[j];

      // ---- P: C layout -> bf16 LDS -> A layout (16x64) -------------------
#pragma unroll
      for (int t = 0; t < 4; ++t)
#pragma unroll
        for (int j = 0; j < 8; ++j)
          my[(j + khalf) * TROW_ + t * 16 + m16] = (__bf16)sc[t][j];

      const v16bf pa0 = load_frag(my + m16 * TROW_ + khalf);       // keys 0..31
      const v16bf pa1 = load_frag(my + m16 * TROW_ + 32 + khalf);  // keys 32..63

      // ---- O += P * V (K = 64 keys, 4 channel tiles of 16) ---------------
#pragma unroll
      for (int c = 0; c < 4; ++c) {
        const __bf16* vrow = &vbuf[bi][(c * 16 + m16) * TROW_];
        oacc[c] = wmma_bf16(pa0, load_frag(vrow + khalf), oacc[c]);
        oacc[c] = wmma_bf16(pa1, load_frag(vrow + 32 + khalf), oacc[c]);
      }
    }

    stage_fence();  // staging of next buffer done; current reads done
  }

  // ---- normalize + store in [B,S,H*dk] so O-projection consumes directly --
  float rinv[8];
#pragma unroll
  for (int j = 0; j < 8; ++j) rinv[j] = 1.0f / lstat[j];
#pragma unroll
  for (int c = 0; c < 4; ++c) {
    const int ch = c * 16 + m16;
#pragma unroll
    for (int j = 0; j < 8; ++j) {
      const int srow = qBase + j + khalf;
      O[(size_t)(b * S_ + srow) * D_ + h * DK_ + ch] =
          (__bf16)(oacc[c][j] * rinv[j]);
    }
  }
}

// ---------------------------------------------------------------------------
// Host launcher
// ---------------------------------------------------------------------------
extern "C" void kernel_launch(void* const* d_in, const int* in_sizes, int n_in,
                              void* d_out, int out_size, void* d_ws,
                              size_t ws_size, hipStream_t stream) {
  (void)in_sizes; (void)n_in; (void)out_size; (void)ws_size;

  const float* x  = (const float*)d_in[0];
  const float* Wq = (const float*)d_in[1];
  const float* bq = (const float*)d_in[2];
  const float* Wk = (const float*)d_in[3];
  const float* bk = (const float*)d_in[4];
  const float* Wv = (const float*)d_in[5];
  const float* bv = (const float*)d_in[6];
  const float* Wo = (const float*)d_in[7];
  const float* bo = (const float*)d_in[8];
  float* out = (float*)d_out;

  const size_t E  = (size_t)B_ * S_ * D_;  // 4,194,304 activations
  const size_t WE = (size_t)D_ * D_;       // 1,048,576 per weight
  char* ws = (char*)d_ws;
  float*  Qf  = (float*)ws;  ws += E * 4;
  float*  Kf  = (float*)ws;  ws += E * 4;
  __bf16* Qb  = (__bf16*)ws; ws += E * 2;
  __bf16* Kb  = (__bf16*)ws; ws += E * 2;
  __bf16* Vt  = (__bf16*)ws; ws += E * 2;  // [B*H][dk][S]
  __bf16* Ab  = (__bf16*)ws; ws += E * 2;
  __bf16* xb  = (__bf16*)ws; ws += E * 2;
  __bf16* Wqb = (__bf16*)ws; ws += WE * 2;
  __bf16* Wkb = (__bf16*)ws; ws += WE * 2;
  __bf16* Wvb = (__bf16*)ws; ws += WE * 2;
  __bf16* Wob = (__bf16*)ws; ws += WE * 2;

  dim3 blk(256);

  // Pre-convert operands to bf16 (one-time; GEMM loops stay pure b128+WMMA).
  f32_to_bf16_kernel<<<(int)(E / 4 / 256), blk, 0, stream>>>(x, xb, (int)(E / 4));
  f32_to_bf16_kernel<<<(int)(WE / 4 / 256), blk, 0, stream>>>(Wq, Wqb, (int)(WE / 4));
  f32_to_bf16_kernel<<<(int)(WE / 4 / 256), blk, 0, stream>>>(Wk, Wkb, (int)(WE / 4));
  f32_to_bf16_kernel<<<(int)(WE / 4 / 256), blk, 0, stream>>>(Wv, Wvb, (int)(WE / 4));
  f32_to_bf16_kernel<<<(int)(WE / 4 / 256), blk, 0, stream>>>(Wo, Wob, (int)(WE / 4));

  const int M = B_ * S_, N = D_, Kd = D_;
  const int totalWaves = (M / 32) * (N / 64);  // 2048
  const int gemmBlocks = totalWaves / 8;       // 256 blocks of 8 waves

  // QKV projections; V written directly transposed+bf16.
  gemm_bias_kernel<float, false><<<gemmBlocks, blk, 0, stream>>>(xb, Wqb, bq, Qf, M, N, Kd);
  gemm_bias_kernel<float, false><<<gemmBlocks, blk, 0, stream>>>(xb, Wkb, bk, Kf, M, N, Kd);
  gemm_bias_kernel<__bf16, true><<<gemmBlocks, blk, 0, stream>>>(xb, Wvb, bv, Vt, M, N, Kd);

  // RoPE + bf16 conversion for Q (pre-scaled) and K.
  rope_convert_kernel<<<(int)(E / 256), blk, 0, stream>>>(Qf, Kf, Qb, Kb);

  // Flash attention: grid.x = q-tiles / 8 waves, grid.y = B*H.
  dim3 agrid(S_ / 16 / 8, B_ * H_);
  flash_attn_kernel<<<agrid, blk, 0, stream>>>(Qb, Kb, Vt, Ab);

  // Output projection.
  gemm_bias_kernel<float, false><<<gemmBlocks, blk, 0, stream>>>(Ab, Wob, bo, out, M, N, Kd);
}
